// AttentionCircuit_69904887709890
// MI455X (gfx1250) — compile-verified, matrix operand order checked by reading, not run on hardware
//
#include <hip/hip_runtime.h>
#include <hip/hip_bf16.h>

// ---------------------------------------------------------------------------
// AttentionCircuit on gfx1250 (MI455X)
//  - all contractions via v_wmma_f32_16x16x32_f16 (f32 accumulate)
//  - GEMM tiles staged with GLOBAL_LOAD_ASYNC_TO_LDS_B128, double-buffered,
//    retired with s_wait_asynccnt (ASYNCcnt)
//  - weights pre-transposed once so both GEMM operands stage as plain rows
// B=2 S=2048 D=1024 N=16 R=256 H=16 DH=64;  BS = 4096;  NR = 4096
// ---------------------------------------------------------------------------

typedef __attribute__((ext_vector_type(16))) _Float16 v16h;
typedef __attribute__((ext_vector_type(8)))  float    v8f;

#define WMMA_F16(a, b, c) \
  __builtin_amdgcn_wmma_f32_16x16x32_f16(false, (a), false, (b), (short)0, (c), false, false)

// Async copy of 16 bytes (8 halfs) global -> LDS; tracked by ASYNCcnt.
__device__ __forceinline__ void async_b128(_Float16* lds, const _Float16* g) {
  uint32_t l = (uint32_t)(uintptr_t)lds;                 // low 32 bits of a generic
  unsigned long long ga = (unsigned long long)(uintptr_t)g;  // LDS pointer = LDS offset
  asm volatile("global_load_async_to_lds_b128 %0, %1, off" :: "v"(l), "v"(ga) : "memory");
}

template <int N> struct WaitAsync;
template <> struct WaitAsync<0> { __device__ static void run() { asm volatile("s_wait_asynccnt 0x0" ::: "memory"); } };
template <> struct WaitAsync<2> { __device__ static void run() { asm volatile("s_wait_asynccnt 0x2" ::: "memory"); } };
template <> struct WaitAsync<4> { __device__ static void run() { asm volatile("s_wait_asynccnt 0x4" ::: "memory"); } };

// Load a 16x32 f16 fragment from an LDS tile stored row-major (stride in halfs).
// ISA 7.12.2, 16-bit A 16x32: lane(0..15) = row; packed pair j covers K=2j (j<4)
// or 16+2(j-4) (j>=4); lanes 16..31 shifted +8 in K. B uses the same pattern
// when its tile is stored N-major ("row" = output column).
__device__ __forceinline__ v16h load_frag(const _Float16* base, int ln, int hf, int stride) {
  v16h f;
#pragma unroll
  for (int j = 0; j < 8; ++j) {
    int k = ((j < 4) ? (2 * j) : (8 + 2 * j)) + hf * 8;
    f[2 * j]     = base[ln * stride + k];
    f[2 * j + 1] = base[ln * stride + k + 1];
  }
  return f;
}

// ------------------------- tile staging (tag dispatch) ---------------------
template <bool P> struct PlainTag {};

// Plain operand: rows contiguous in K -> two async b128 per thread (64x32 tile).
template <typename F>
__device__ __forceinline__ void stage_tile(PlainTag<true>, const F& f, _Float16* s,
                                           int tid, int base, int k0) {
  int r = tid >> 1, ch = (tid & 1) * 16;
  const _Float16* g = f.rowptr(base + r, k0 + ch);
  _Float16* d = s + r * 40 + ch;
  async_b128(d, g);
  async_b128(d + 8, g + 8);
}
// Computed operand: manual element staging through VGPRs (DS stores).
template <typename F>
__device__ __forceinline__ void stage_tile(PlainTag<false>, const F& f, _Float16* s,
                                           int tid, int base, int k0) {
  for (int idx = tid; idx < 64 * 32; idx += 128) {
    int r = idx >> 5, c = idx & 31;
    s[r * 40 + c] = f(base + r, k0 + c);
  }
}

// ---------------------------------------------------------------------------
// Tiled WMMA GEMM: C[M,N] = A[M,K] * Bt[N,K]^T, double-buffered async staging.
// 128 threads = 4 waves; block tile 64x64, K-step 32; wave tile 32x32 (2x2 wmma).
// FA(row,k)->f16; FBT(col,k)->f16 (B pre-transposed); FC(row,col,acc).
// M,N multiples of 64; K multiple of 32.
// ---------------------------------------------------------------------------
template <typename FA, typename FBT, typename FC>
__global__ void __launch_bounds__(128)
gemm_wmma_kernel(FA fa, FBT fbt, FC fc, int M, int N, int K) {
  __shared__ _Float16 sA[2][64 * 40];  // [m][k], padded stride 40
  __shared__ _Float16 sB[2][64 * 40];  // [n][k]
  const int tid  = threadIdx.x;
  const int wave = tid >> 5;
  const int lane = tid & 31;
  const int hf   = (lane >> 4) & 1;
  const int ln   = lane & 15;
  const int wm   = (wave & 1) * 32;
  const int wn   = (wave >> 1) * 32;
  const int row0 = blockIdx.y * 64;
  const int col0 = blockIdx.x * 64;
  constexpr int PEND = (FA::plain ? 2 : 0) + (FBT::plain ? 2 : 0);

  v8f acc[2][2];
#pragma unroll
  for (int i = 0; i < 2; ++i)
#pragma unroll
    for (int j = 0; j < 2; ++j) acc[i][j] = {};

  // prologue: stage tile 0 into buffer 0
  stage_tile(PlainTag<FA::plain>{}, fa, sA[0], tid, row0, 0);
  stage_tile(PlainTag<FBT::plain>{}, fbt, sB[0], tid, col0, 0);

  int cur = 0;
  for (int k0 = 0; k0 < K; k0 += 32) {
    const int nxt = cur ^ 1;
    const bool has_next = (k0 + 32 < K);
    __syncthreads();  // readers of buffer `nxt` (tile k-1) are done
    if (has_next) {
      stage_tile(PlainTag<FA::plain>{}, fa, sA[nxt], tid, row0, k0 + 32);
      stage_tile(PlainTag<FBT::plain>{}, fbt, sB[nxt], tid, col0, k0 + 32);
      WaitAsync<PEND>::run();  // retire tile k (older asyncs complete in order)
    } else {
      WaitAsync<0>::run();
    }
    __syncthreads();  // tile k visible to all waves

    v16h aF[2], bF[2];
#pragma unroll
    for (int i = 0; i < 2; ++i) aF[i] = load_frag(&sA[cur][(wm + 16 * i) * 40], ln, hf, 40);
#pragma unroll
    for (int j = 0; j < 2; ++j) bF[j] = load_frag(&sB[cur][(wn + 16 * j) * 40], ln, hf, 40);
#pragma unroll
    for (int i = 0; i < 2; ++i)
#pragma unroll
      for (int j = 0; j < 2; ++j) acc[i][j] = WMMA_F16(aF[i], bF[j], acc[i][j]);
    cur = nxt;
  }
  // C/D layout: element e -> row = e + 8*hf (within 16-row tile), col = ln
#pragma unroll
  for (int i = 0; i < 2; ++i)
#pragma unroll
    for (int j = 0; j < 2; ++j)
#pragma unroll
      for (int e = 0; e < 8; ++e) {
        int r = row0 + wm + 16 * i + e + 8 * hf;
        int c = col0 + wn + 16 * j + ln;
        fc(r, c, acc[i][j][e]);
      }
}

// ------------------------- operand functors --------------------------------
struct AHalf {  // plain f16 A, row-major [M][K]
  const _Float16* p; int ld;
  static constexpr bool plain = true;
  __device__ const _Float16* rowptr(int r, int k) const { return p + (size_t)r * ld + k; }
  __device__ _Float16 operator()(int r, int k) const { return p[(size_t)r * ld + k]; }
};
struct BTHalf {  // plain f16 B, PRE-TRANSPOSED [N][K]
  const _Float16* p; int ld;  // ld == K
  static constexpr bool plain = true;
  __device__ const _Float16* rowptr(int c, int k) const { return p + (size_t)c * ld + k; }
  __device__ _Float16 operator()(int c, int k) const { return p[(size_t)c * ld + k]; }
};
// Restore-GEMM A: G[bs,k] = h[bs, k&255] * w[bs, k>>8] (fused, never materialized)
struct ARestore {
  const _Float16* h; const float* w;
  static constexpr bool plain = false;
  __device__ _Float16 operator()(int r, int k) const {
    int n = k >> 8, rr = k & 255;
    return (_Float16)((float)h[(size_t)r * 256 + rr] * w[(size_t)r * 16 + n]);
  }
};
// Final-GEMM A: gather attention output [b,h,s,dh] back to [bs, d]
struct AFromO {
  const _Float16* o;
  static constexpr bool plain = false;
  __device__ _Float16 operator()(int r, int k) const {
    int b = r >> 11, s = r & 2047, hh = k >> 6, dh = k & 63;
    return o[((size_t)(b * 16 + hh) * 2048 + s) * 64 + dh];
  }
};
struct CHalf {  // f16 store, row-major
  _Float16* p; int ld;
  __device__ void operator()(int r, int c, float v) const { p[(size_t)r * ld + c] = (_Float16)v; }
};
struct CFloat {  // f32 store, row-major
  float* p; int ld;
  __device__ void operator()(int r, int c, float v) const { p[(size_t)r * ld + c] = v; }
};
// Restore-GEMM C: scatter [bs,d] -> f16 [b,h,s,dh] attention layout, with scale
struct CScatterQKV {
  _Float16* p; float scale;
  __device__ void operator()(int r, int c, float v) const {
    int b = r >> 11, s = r & 2047, hh = c >> 6, dh = c & 63;
    p[((size_t)(b * 16 + hh) * 2048 + s) * 64 + dh] = (_Float16)(v * scale);
  }
};

// ------------------------- prep kernels ------------------------------------
__global__ void cvt_f32_f16_kernel(const float* __restrict__ src, _Float16* __restrict__ dst, int n) {
  int i = blockIdx.x * blockDim.x + threadIdx.x;
  if (i < n) dst[i] = (_Float16)src[i];
}

// Ft[(n*256+r)*1024 + d] = f[n,d,r]  : feature weights -> [NR][D] (B^T layout)
__global__ void packFT_kernel(const float* __restrict__ f, _Float16* __restrict__ Ft) {
  int idx = blockIdx.x * blockDim.x + threadIdx.x;  // 4096*1024
  if (idx >= 4096 * 1024) return;
  int c = idx >> 10, d = idx & 1023;
  int n = c >> 8, r = c & 255;
  Ft[idx] = (_Float16)f[((size_t)n * 1024 + d) * 256 + r];
}

// dst[c][k] = (f16) src[k][c] : transpose+convert; K = 1<<kl, C columns in src
__global__ void cvtT_kernel(const float* __restrict__ src, _Float16* __restrict__ dst,
                            int kl, int C, int n) {
  int idx = blockIdx.x * blockDim.x + threadIdx.x;
  if (idx >= n) return;
  int k = idx & ((1 << kl) - 1), c = idx >> kl;
  dst[idx] = (_Float16)src[(size_t)k * C + c];
}

// h[bs,r] = sum_n Y[bs, n*256+r] * w[bs,n]  (two weight sets share one Y read)
__global__ void combine_kernel(const _Float16* __restrict__ Y,
                               const float* __restrict__ wA, const float* __restrict__ wB,
                               _Float16* __restrict__ hA, _Float16* __restrict__ hB) {
  int idx = blockIdx.x * blockDim.x + threadIdx.x;  // 4096*256
  if (idx >= 4096 * 256) return;
  int bs = idx >> 8, r = idx & 255;
  const _Float16* yrow = Y + (size_t)bs * 4096 + r;
  float a = 0.f, b = 0.f;
#pragma unroll
  for (int n = 0; n < 16; ++n) {
    float y = (float)yrow[(size_t)n * 256];
    a += y * wA[bs * 16 + n];
    if (wB) b += y * wB[bs * 16 + n];
  }
  hA[(size_t)bs * 256 + r] = (_Float16)a;
  if (hB) hB[(size_t)bs * 256 + r] = (_Float16)b;
}

// ---------------------------------------------------------------------------
// Flash attention, causal. One block = 4 waves = 64 query rows of one (b,h).
// QK^T and P@V on WMMA; online softmax with per-16-lane shfl reductions.
// Q,K tiles staged via async-to-LDS; V transposed manually. Q pre-scaled.
// ---------------------------------------------------------------------------
__global__ void __launch_bounds__(128)
attn_kernel(const _Float16* __restrict__ Qb, const _Float16* __restrict__ Kb,
            const _Float16* __restrict__ Vb, _Float16* __restrict__ Ob) {
  __shared__ _Float16 sQ[64 * 72];     // [q][dh]
  __shared__ _Float16 sK[64 * 72];     // [key][dh]  (N-major B tile for QK^T)
  __shared__ _Float16 sVt[64 * 72];    // [dh][key]  (N-major B tile for P@V)
  __shared__ _Float16 sP[4][16 * 72];  // per-wave P strip [q_local][key_local]

  const int tid  = threadIdx.x;
  const int wave = tid >> 5;
  const int lane = tid & 31;
  const int hf   = (lane >> 4) & 1;
  const int ln   = lane & 15;
  const int bh   = blockIdx.x >> 5;
  const int qblk = blockIdx.x & 31;
  const int q0   = qblk * 64;

  const _Float16* Qh = Qb + (size_t)bh * 2048 * 64;
  const _Float16* Kh = Kb + (size_t)bh * 2048 * 64;
  const _Float16* Vh = Vb + (size_t)bh * 2048 * 64;
  _Float16*       Oh = Ob + (size_t)bh * 2048 * 64;

  {  // async-stage Q tile (retired by first in-loop wait)
    int r = tid >> 1, ch = (tid & 1) * 32;
    const _Float16* g = Qh + (size_t)(q0 + r) * 64 + ch;
    _Float16* s = &sQ[r * 72 + ch];
#pragma unroll
    for (int u = 0; u < 4; ++u) async_b128(s + u * 8, g + u * 8);
  }

  float m_i[8], l_i[8];
  v8f acc[4];
#pragma unroll
  for (int i = 0; i < 8; ++i) { m_i[i] = -3.0e38f; l_i[i] = 0.f; }
#pragma unroll
  for (int t = 0; t < 4; ++t) acc[t] = {};

  for (int j0 = 0; j0 <= q0; j0 += 64) {
    __syncthreads();
    {  // async-stage K tile
      int r = tid >> 1, ch = (tid & 1) * 32;
      const _Float16* g = Kh + (size_t)(j0 + r) * 64 + ch;
      _Float16* s = &sK[r * 72 + ch];
#pragma unroll
      for (int u = 0; u < 4; ++u) async_b128(s + u * 8, g + u * 8);
    }
    for (int idx = tid; idx < 64 * 64; idx += 128) {  // V transpose via VGPRs
      int r = idx >> 6, c = idx & 63;
      sVt[c * 72 + r] = Vh[(size_t)(j0 + r) * 64 + c];
    }
    WaitAsync<0>::run();
    __syncthreads();

    // S strip = Q(16 wave rows) @ K^T(64 keys); K-dim = DH = 64 = 2 steps
    v8f sacc[4];
#pragma unroll
    for (int t = 0; t < 4; ++t) sacc[t] = {};
#pragma unroll
    for (int kk = 0; kk < 2; ++kk) {
      v16h aQ = load_frag(sQ + (wave * 16) * 72 + kk * 32, ln, hf, 72);
#pragma unroll
      for (int t = 0; t < 4; ++t) {
        v16h bK = load_frag(sK + (t * 16) * 72 + kk * 32, ln, hf, 72);
        sacc[t] = WMMA_F16(aQ, bK, sacc[t]);
      }
    }

    // online softmax update (row = e + 8*hf within this wave's 16-row strip)
#pragma unroll
    for (int i = 0; i < 8; ++i) {
      int row_l = i + 8 * hf;
      int grow  = q0 + wave * 16 + row_l;
      float vt[4];
#pragma unroll
      for (int t = 0; t < 4; ++t) {
        float v = sacc[t][i];
        int gcol = j0 + t * 16 + ln;
        if (gcol > grow) v = -3.0e38f;  // causal mask
        vt[t] = v;
      }
      float mx = fmaxf(fmaxf(vt[0], vt[1]), fmaxf(vt[2], vt[3]));
#pragma unroll
      for (int off = 1; off < 16; off <<= 1) mx = fmaxf(mx, __shfl_xor(mx, off, 32));
      float m_new = fmaxf(m_i[i], mx);
      float sc    = __expf(m_i[i] - m_new);
      float rsum  = 0.f;
#pragma unroll
      for (int t = 0; t < 4; ++t) {
        float p = __expf(vt[t] - m_new);
        rsum += p;
        sP[wave][row_l * 72 + t * 16 + ln] = (_Float16)p;
      }
#pragma unroll
      for (int off = 1; off < 16; off <<= 1) rsum += __shfl_xor(rsum, off, 32);
      l_i[i] = l_i[i] * sc + rsum;
      m_i[i] = m_new;
#pragma unroll
      for (int t = 0; t < 4; ++t) acc[t][i] *= sc;
    }
    __syncthreads();

    // O += P(16x64) @ V(64 keys x 64 dh); K-dim = 64 keys = 2 steps
#pragma unroll
    for (int kk = 0; kk < 2; ++kk) {
      v16h aP = load_frag(&sP[wave][0] + kk * 32, ln, hf, 72);
#pragma unroll
      for (int t = 0; t < 4; ++t) {
        v16h bV = load_frag(sVt + (t * 16) * 72 + kk * 32, ln, hf, 72);
        acc[t] = WMMA_F16(aP, bV, acc[t]);
      }
    }
  }

#pragma unroll
  for (int i = 0; i < 8; ++i) {
    int row_l = i + 8 * hf;
    float inv = 1.0f / l_i[i];
#pragma unroll
    for (int t = 0; t < 4; ++t)
      Oh[(size_t)(q0 + wave * 16 + row_l) * 64 + t * 16 + ln] = (_Float16)(acc[t][i] * inv);
  }
}

// ---------------------------------------------------------------------------
extern "C" void kernel_launch(void* const* d_in, const int* in_sizes, int n_in,
                              void* d_out, int out_size, void* d_ws, size_t ws_size,
                              hipStream_t stream) {
  (void)in_sizes; (void)n_in; (void)out_size; (void)ws_size;
  const float* x    = (const float*)d_in[0];   // [2,2048,1024]
  const float* f_qk = (const float*)d_in[1];   // [16,1024,256]
  const float* f_v  = (const float*)d_in[2];
  const float* r_qk = (const float*)d_in[3];   // [16,256,1024] == [4096,1024]
  const float* r_v  = (const float*)d_in[4];
  const float* wfqQ = (const float*)d_in[5];   // [4096,16]
  const float* wfqK = (const float*)d_in[6];
  const float* wfv  = (const float*)d_in[7];
  const float* wrqQ = (const float*)d_in[8];
  const float* wrqK = (const float*)d_in[9];
  const float* wrv  = (const float*)d_in[10];
  const float* W_O  = (const float*)d_in[11];  // [1024,1024]
  float* out = (float*)d_out;                  // [2,2048,1024] f32

  // workspace carve-up (halfs); total = 56M halfs = 112 MB
  _Float16* w = (_Float16*)d_ws;
  const size_t M4 = (size_t)4096 * 1024;  // 4M
  _Float16* xh   = w;  w += M4;                    // x f16 [4096][1024]
  _Float16* FqkT = w;  w += M4;                    // [4096][1024]  (B^T)
  _Float16* FvT  = w;  w += M4;
  _Float16* RqkT = w;  w += M4;                    // [1024][4096]  (B^T)
  _Float16* RvT  = w;  w += M4;
  _Float16* WOT  = w;  w += (size_t)1024 * 1024;   // [1024][1024]  (B^T)
  _Float16* Y    = w;  w += (size_t)4096 * 4096;   // 16M
  _Float16* hq   = w;  w += (size_t)4096 * 256;
  _Float16* hk   = w;  w += (size_t)4096 * 256;
  _Float16* hv   = w;  w += (size_t)4096 * 256;
  _Float16* Qb   = w;  w += M4;                    // [b,h,s,dh] f16
  _Float16* Kb   = w;  w += M4;
  _Float16* Vb   = w;  w += M4;
  _Float16* Ob   = w;  w += M4;

  // 1) precision conversion / packing (weights emitted pre-transposed [N][K])
  cvt_f32_f16_kernel<<<(int)(M4 / 256), 256, 0, stream>>>(x, xh, (int)M4);
  packFT_kernel<<<(int)(M4 / 256), 256, 0, stream>>>(f_qk, FqkT);
  packFT_kernel<<<(int)(M4 / 256), 256, 0, stream>>>(f_v, FvT);
  cvtT_kernel<<<(int)(M4 / 256), 256, 0, stream>>>(r_qk, RqkT, 12, 1024, (int)M4);
  cvtT_kernel<<<(int)(M4 / 256), 256, 0, stream>>>(r_v, RvT, 12, 1024, (int)M4);
  cvtT_kernel<<<(1024 * 1024) / 256, 256, 0, stream>>>(W_O, WOT, 10, 1024, 1024 * 1024);

  dim3 gFeat(4096 / 64, 4096 / 64);   // (N/64, M/64)
  dim3 gRest(1024 / 64, 4096 / 64);
  // 2) Y = x @ F_qk ; combine -> h_q, h_k (shared feature matmul for Q and K)
  gemm_wmma_kernel<<<gFeat, 128, 0, stream>>>(AHalf{xh, 1024}, BTHalf{FqkT, 1024},
                                              CHalf{Y, 4096}, 4096, 4096, 1024);
  combine_kernel<<<4096, 256, 0, stream>>>(Y, wfqQ, wfqK, hq, hk);
  // 3) Y = x @ F_v ; combine -> h_v
  gemm_wmma_kernel<<<gFeat, 128, 0, stream>>>(AHalf{xh, 1024}, BTHalf{FvT, 1024},
                                              CHalf{Y, 4096}, 4096, 4096, 1024);
  combine_kernel<<<4096, 256, 0, stream>>>(Y, wfv, (const float*)nullptr, hv, (_Float16*)nullptr);

  // 4) restore GEMMs (weight-fused A), scatter into [b,h,s,dh]; Q pre-scaled 1/8
  gemm_wmma_kernel<<<gRest, 128, 0, stream>>>(ARestore{hq, wrqQ}, BTHalf{RqkT, 4096},
                                              CScatterQKV{Qb, 0.125f}, 4096, 1024, 4096);
  gemm_wmma_kernel<<<gRest, 128, 0, stream>>>(ARestore{hk, wrqK}, BTHalf{RqkT, 4096},
                                              CScatterQKV{Kb, 1.0f}, 4096, 1024, 4096);
  gemm_wmma_kernel<<<gRest, 128, 0, stream>>>(ARestore{hv, wrv}, BTHalf{RvT, 4096},
                                              CScatterQKV{Vb, 1.0f}, 4096, 1024, 4096);

  // 5) causal flash attention: 32 (b,h) x 32 query-blocks of 64
  attn_kernel<<<32 * 32, 128, 0, stream>>>(Qb, Kb, Vb, Ob);

  // 6) out = O(reordered to [bs,d]) @ W_O -> f32
  gemm_wmma_kernel<<<gRest, 128, 0, stream>>>(AFromO{Ob}, BTHalf{WOT, 1024},
                                              CFloat{out, 1024}, 4096, 1024, 1024);
}
